// GNN_14113262535394
// MI455X (gfx1250) — compile-verified, matrix operand order checked by reading, not run on hardware
//
#include <hip/hip_runtime.h>
#include <math.h>

typedef __bf16 bf16_t;
typedef __attribute__((ext_vector_type(16))) __bf16 v16bf;
typedef __attribute__((ext_vector_type(8)))  float  v8f;

// ---------------------------------------------------------------------------
// utility fills
// ---------------------------------------------------------------------------
__global__ void fill_f32(float* __restrict__ p, float v, int n) {
    int t = blockIdx.x * blockDim.x + threadIdx.x;
    if (t < n) p[t] = v;
}

// deg starts at 1.0 (self loop); add 1 per incoming edge
__global__ void degree_kernel(const int* __restrict__ dst, float* __restrict__ deg, int E) {
    int t = blockIdx.x * blockDim.x + threadIdx.x;
    if (t < E) atomicAdd(&deg[dst[t]], 1.0f);
}

__global__ void rsqrt_inplace(float* __restrict__ d, int n) {
    int t = blockIdx.x * blockDim.x + threadIdx.x;
    if (t < n) d[t] = rsqrtf(d[t]);
}

// ---------------------------------------------------------------------------
// pack W (K x Ncols, f32 row-major) into per-lane WMMA B-fragment bf16 layout:
// fragment (tk,tn): 32 lanes x 16 bf16, lane L element j holds
//   W[tk*32 + (L/16)*16 + j][tn*16 + (L%16)]
// stored contiguously: Wp[((tk*nT+tn)*32 + L)*16 + j]
// ---------------------------------------------------------------------------
__global__ void pack_w_bf16(const float* __restrict__ W, bf16_t* __restrict__ Wp,
                            int K, int Ncols) {
    int t = blockIdx.x * blockDim.x + threadIdx.x;
    int total = K * Ncols;
    if (t >= total) return;
    int nT   = Ncols >> 4;
    int j    = t & 15;
    int lane = (t >> 4) & 31;
    int frag = t >> 9;           // tk*nT + tn
    int tn   = frag % nT;
    int tk   = frag / nT;
    int k    = tk * 32 + (lane >> 4) * 16 + j;
    int n    = tn * 16 + (lane & 15);
    Wp[t] = (bf16_t)W[(size_t)k * Ncols + n];
}

// ---------------------------------------------------------------------------
// WMMA GEMM: C[M,Ncols] = A[M,K] (f32 -> bf16 in regs) * Wp (pre-packed bf16)
// block = 256 threads = 8 waves arranged 4(M) x 2(N); block tile 128x128
// wave tile 32x64 = 2x4 WMMA accumulators; K fully unrolled (KSTEPS steps of 32)
// Out-of-range rows are CLAMPED for loads (garbage compute) and guarded only
// at the store -> no EXEC-mask manipulation in the hot loop.
// ---------------------------------------------------------------------------
#define CVT4(dstv, base, f)                              \
    dstv[(base) + 0] = (bf16_t)(f).x;                    \
    dstv[(base) + 1] = (bf16_t)(f).y;                    \
    dstv[(base) + 2] = (bf16_t)(f).z;                    \
    dstv[(base) + 3] = (bf16_t)(f).w;

template <int KSTEPS>
__global__ __launch_bounds__(256)
void gemm_bf16_wmma(const float* __restrict__ A, const bf16_t* __restrict__ Bp,
                    float* __restrict__ C, int M, int Ncols) {
    const int K     = KSTEPS * 32;
    const int lane  = threadIdx.x & 31;
    const int wave  = threadIdx.x >> 5;
    const int waveM = wave & 3;
    const int waveN = wave >> 2;
    const int m0    = blockIdx.x * 128 + waveM * 32;
    const int n0    = blockIdx.y * 128 + waveN * 64;
    const int nT    = Ncols >> 4;
    const int lmod  = lane & 15;
    const int lhi   = lane >> 4;     // 0 or 1
    const int tn0   = n0 >> 4;

    v8f acc[2][4];
#pragma unroll
    for (int mi = 0; mi < 2; ++mi)
#pragma unroll
        for (int ni = 0; ni < 4; ++ni)
#pragma unroll
            for (int q = 0; q < 8; ++q) acc[mi][ni][q] = 0.0f;

    // K-invariant base pointers (rows clamped; stores are guarded instead)
    const float* aptr[2];
#pragma unroll
    for (int mi = 0; mi < 2; ++mi) {
        int row = m0 + mi * 16 + lmod;
        if (row >= M) row = M - 1;
        aptr[mi] = A + (size_t)row * K + lhi * 16;
    }
    const bf16_t* bptr   = Bp + (size_t)tn0 * 512 + (size_t)lane * 16;
    const size_t  bkstep = (size_t)nT * 512;   // elements per k-step

#pragma unroll
    for (int ks = 0; ks < KSTEPS; ++ks) {
        v16bf afrag[2];
#pragma unroll
        for (int mi = 0; mi < 2; ++mi) {
            const float4* ap = (const float4*)(aptr[mi] + ks * 32);
            float4 f0 = ap[0];
            float4 f1 = ap[1];
            float4 f2 = ap[2];
            float4 f3 = ap[3];
            CVT4(afrag[mi], 0, f0);
            CVT4(afrag[mi], 4, f1);
            CVT4(afrag[mi], 8, f2);
            CVT4(afrag[mi], 12, f3);
        }
        v16bf bfrag[4];
#pragma unroll
        for (int ni = 0; ni < 4; ++ni) {
            bfrag[ni] = *(const v16bf*)(bptr + ks * bkstep + (size_t)ni * 512);
        }
#pragma unroll
        for (int mi = 0; mi < 2; ++mi)
#pragma unroll
            for (int ni = 0; ni < 4; ++ni)
                acc[mi][ni] = __builtin_amdgcn_wmma_f32_16x16x32_bf16(
                    false, afrag[mi], false, bfrag[ni],
                    (short)0, acc[mi][ni], false, false);
    }

    // store: lane L, acc element v -> row m0+mi*16 + v + 8*(L/16), col tile + L%16
#pragma unroll
    for (int mi = 0; mi < 2; ++mi) {
#pragma unroll
        for (int v = 0; v < 8; ++v) {
            int row = m0 + mi * 16 + v + 8 * lhi;
            if (row < M) {
#pragma unroll
                for (int ni = 0; ni < 4; ++ni) {
                    C[(size_t)row * Ncols + n0 + ni * 16 + lmod] = acc[mi][ni][v];
                }
            }
        }
    }
}

// ---------------------------------------------------------------------------
// edge scatter: AGG[dst] += H[src] * dis[src]*dis[dst]   (float4 chunks, f32 atomics)
// ---------------------------------------------------------------------------
__global__ void scatter_agg(const float* __restrict__ H, const int* __restrict__ src,
                            const int* __restrict__ dst, const float* __restrict__ dis,
                            float* AGG, int E, int F, int lc /* log2(F/4) */) {
    int chunks = 1 << lc;
    int t = threadIdx.x;
    int e = blockIdx.x * (blockDim.x >> lc) + (t >> lc);
    int c = t & (chunks - 1);
    if (e >= E) return;
    int s = src[e], d = dst[e];
    float w = dis[s] * dis[d];
    float4 h = ((const float4*)(H + (size_t)s * F))[c];
    float* out = AGG + (size_t)d * F + c * 4;
    atomicAdd(out + 0, h.x * w);
    atomicAdd(out + 1, h.y * w);
    atomicAdd(out + 2, h.z * w);
    atomicAdd(out + 3, h.w * w);
}

// ---------------------------------------------------------------------------
// epilogue: X = gelu_exact(AGG + H*dis^2 (self loop) + bias)  (in-place on AGG ok)
// ---------------------------------------------------------------------------
__global__ void epilogue_gelu(const float* AGG, const float* __restrict__ H,
                              const float* __restrict__ dis, const float* __restrict__ bias,
                              float* X, int n, int lF /* log2(F) */) {
    int t = blockIdx.x * blockDim.x + threadIdx.x;
    if (t >= (n << lF)) return;
    int i = t >> lF;
    int c = t & ((1 << lF) - 1);
    float s = dis[i];
    float x = AGG[t] + H[t] * s * s + bias[c];
    X[t] = 0.5f * x * (1.0f + erff(x * 0.70710678118654752f));
}

// ---------------------------------------------------------------------------
// launch
// ---------------------------------------------------------------------------
extern "C" void kernel_launch(void* const* d_in, const int* in_sizes, int n_in,
                              void* d_out, int out_size, void* d_ws, size_t ws_size,
                              hipStream_t stream) {
    const float* node_emb = (const float*)d_in[0];
    const float* W1       = (const float*)d_in[1];
    const float* b1       = (const float*)d_in[2];
    const float* W2       = (const float*)d_in[3];
    const float* b2       = (const float*)d_in[4];
    const int*   ei       = (const int*)d_in[5];

    const int MID   = in_sizes[2];             // 256
    const int OUTF  = in_sizes[4];             // 128
    const int EMB   = in_sizes[1] / MID;       // 256
    const int Nn    = in_sizes[0] / EMB;       // 50000
    const int E     = in_sizes[5] / 2;         // 800000
    const int* src  = ei;
    const int* dst  = ei + E;

    // workspace carve-up
    auto al = [](size_t x) { return (x + 255) & ~(size_t)255; };
    char*   w    = (char*)d_ws;
    size_t  off  = 0;
    float*  dis  = (float*)(w + off);  off += al((size_t)Nn * 4);
    bf16_t* Wp1  = (bf16_t*)(w + off); off += al((size_t)EMB * MID * 2);
    bf16_t* Wp2  = (bf16_t*)(w + off); off += al((size_t)MID * OUTF * 2);
    float*  Hbuf = (float*)(w + off);  off += al((size_t)Nn * MID * 4);   // H1, reused as H2
    float*  Abuf = (float*)(w + off);  off += al((size_t)Nn * MID * 4);   // AGG1 -> X1 (in place)
    float*  Agg2 = (float*)(w + off);  off += al((size_t)Nn * OUTF * 4);
    (void)ws_size; (void)n_in; (void)out_size;

    const int T = 256;
    auto nb = [T](long n) { return (int)((n + T - 1) / T); };
    auto lg2 = [](int x) { return 31 - __builtin_clz(x); };

    // 0) init: deg=1 (self loop), AGG buffers = 0  (every call -> deterministic work)
    fill_f32<<<nb(Nn), T, 0, stream>>>(dis, 1.0f, Nn);
    fill_f32<<<nb((long)Nn * MID), T, 0, stream>>>(Abuf, 0.0f, Nn * MID);
    fill_f32<<<nb((long)Nn * OUTF), T, 0, stream>>>(Agg2, 0.0f, Nn * OUTF);

    // 1) degrees -> dis = rsqrt(deg)
    degree_kernel<<<nb(E), T, 0, stream>>>(dst, dis, E);
    rsqrt_inplace<<<nb(Nn), T, 0, stream>>>(dis, Nn);

    // 2) pack weights into WMMA B-fragment bf16 layout
    pack_w_bf16<<<nb((long)EMB * MID), T, 0, stream>>>(W1, Wp1, EMB, MID);
    pack_w_bf16<<<nb((long)MID * OUTF), T, 0, stream>>>(W2, Wp2, MID, OUTF);

    // 3) layer 1: H1 = node_emb @ W1    (K = EMB = 256 -> 8 k-steps, fully unrolled)
    {
        dim3 grid((Nn + 127) / 128, MID / 128);
        gemm_bf16_wmma<8><<<grid, T, 0, stream>>>(node_emb, Wp1, Hbuf, Nn, MID);
    }
    // 4) aggregate + epilogue (bias + self loop + exact gelu) -> X1 in Abuf
    {
        int lc = lg2(MID >> 2);
        scatter_agg<<<nb((long)E << lc), T, 0, stream>>>(Hbuf, src, dst, dis, Abuf, E, MID, lc);
        epilogue_gelu<<<nb((long)Nn * MID), T, 0, stream>>>(Abuf, Hbuf, dis, b1, Abuf,
                                                           Nn, lg2(MID));
    }
    // 5) layer 2: H2 = X1 @ W2  (into Hbuf; K = MID = 256 -> 8 k-steps)
    {
        dim3 grid((Nn + 127) / 128, OUTF / 128);
        gemm_bf16_wmma<8><<<grid, T, 0, stream>>>(Abuf, Wp2, Hbuf, Nn, OUTF);
    }
    // 6) aggregate + epilogue -> d_out (full X = concat of the two returned slices)
    {
        int lc = lg2(OUTF >> 2);
        scatter_agg<<<nb((long)E << lc), T, 0, stream>>>(Hbuf, src, dst, dis, Agg2, E, OUTF, lc);
        epilogue_gelu<<<nb((long)Nn * OUTF), T, 0, stream>>>(Agg2, Hbuf, dis, b2,
                                                            (float*)d_out, Nn, lg2(OUTF));
    }
}
